// MultiHeadDifferentialAttention_11562051961005
// MI455X (gfx1250) — compile-verified, matrix operand order checked by reading, not run on hardware
//
#include <hip/hip_runtime.h>

// ---------------------------------------------------------------------------
// MultiHeadDifferentialAttention on MI455X (gfx1250), wave32 + WMMA f16.
// Pipeline: cvt/transpose -> QKV WMMA GEMMs (TDM-staged, double-buffered LDS)
// -> flash-style dual-softmax differential attention (WMMA) + LayerNorm
// -> output WMMA GEMM (+bias).
// ---------------------------------------------------------------------------

typedef __attribute__((ext_vector_type(16))) _Float16 v16h;
typedef __attribute__((ext_vector_type(8)))  _Float16 v8h;
typedef __attribute__((ext_vector_type(8)))  float    v8f;
typedef __attribute__((ext_vector_type(4)))  unsigned int v4u;
typedef __attribute__((ext_vector_type(8)))  int      v8i;
typedef __attribute__((ext_vector_type(4)))  int      v4i;

#define DEV __device__ __forceinline__

DEV v8f wmma32(v16h a, v16h b, v8f c) {
  // D = A(16x32 f16) * B(32x16 f16) + C(16x16 f32)
  return __builtin_amdgcn_wmma_f32_16x16x32_f16(
      /*neg_a=*/false, a, /*neg_b=*/false, b,
      /*c_mod=*/(short)0, c, /*reuse_a=*/false, /*reuse_b=*/false);
}

// Load a 16x32 f16 WMMA operand from row-major memory (row stride = ld halves).
// Valid for the A operand (lane = M row) and for the B operand when memory
// holds B transposed (row index = WMMA N column): lanes<16 hold K 0..7 &
// 16..23, lanes>=16 hold 8..15 & 24..31.
DEV v16h load_op16(const _Float16* p, int ld) {
  const int lane = threadIdx.x & 31;
  const int row  = lane & 15;
  const int k0   = (lane >> 4) << 3;          // 0 or 8
  const _Float16* r = p + (size_t)row * ld + k0;
  v8h lo = *(const v8h*)(r);
  v8h hi = *(const v8h*)(r + 16);
  return __builtin_shufflevector(lo, hi, 0,1,2,3,4,5,6,7,8,9,10,11,12,13,14,15);
}

// Row-wise reductions across the 16-lane groups of the C/D layout.
DEV float redmax16(float v) {
#pragma unroll
  for (int m = 1; m < 16; m <<= 1) v = fmaxf(v, __shfl_xor(v, m, 32));
  return v;
}
DEV float redsum16(float v) {
#pragma unroll
  for (int m = 1; m < 16; m <<= 1) v += __shfl_xor(v, m, 32);
  return v;
}

// ---------------------------------------------------------------------------
// TDM: async 2D tile load (rows x 32 halves) from global to LDS.
// D# per CDNA5 ISA ch.8: group0 = {count, lds_addr, global_addr, type=2},
// group1 = {data_size=2B, tensor dims, tile dims 32 x rows, dim0 stride}.
// ---------------------------------------------------------------------------
DEV void tdm_load_tile_b16(unsigned lds_off, const void* gaddr, int ld_halves,
                           int rows) {
  unsigned long long ga = (unsigned long long)(size_t)gaddr;
  v4u g0 = {};
  g0[0] = 1u;                                            // count=1, user mode
  g0[1] = lds_off;                                       // lds_addr
  g0[2] = (unsigned)(ga & 0xffffffffu);                  // global_addr[31:0]
  g0[3] = (unsigned)((ga >> 32) & 0x01ffffffu) | (2u << 30);  // [56:32] | type=2
  v8i g1 = {};
  g1[0] = (int)(1u << 16);                               // data_size = 2 bytes
  g1[1] = (int)(((unsigned)ld_halves & 0xffffu) << 16);  // tensor_dim0[15:0]
  g1[2] = (int)(((unsigned)ld_halves >> 16) & 0xffffu);  // tensor_dim0[31:16]
  g1[3] = (int)((16u) | (32u << 16));                    // tensor_dim1=1<<20 hi | tile_dim0=32
  g1[4] = (int)((unsigned)rows & 0xffffu);               // tile_dim1 = rows
  g1[5] = (int)(unsigned)ld_halves;                      // tensor_dim0_stride[31:0]
  g1[6] = 0;                                             // stride hi | dim1_stride lo
  g1[7] = 0;
  v4i gz = {};
#if __clang_major__ >= 23
  v8i gz8 = {};
  __builtin_amdgcn_tensor_load_to_lds(g0, g1, gz, gz, gz8, 0);
#else
  __builtin_amdgcn_tensor_load_to_lds(g0, g1, gz, gz, 0);
#endif
}

// ---------------------------------------------------------------------------
// Pre-pass kernels: f32 -> f16 convert, and f32 W(KxN) -> f16 Wt(NxK).
// ---------------------------------------------------------------------------
__global__ void k_cvt_f32_f16(const float* __restrict__ in,
                              _Float16* __restrict__ out, int n) {
  int i = blockIdx.x * blockDim.x + threadIdx.x;
  int stride = gridDim.x * blockDim.x;
  for (; i < n; i += stride) out[i] = (_Float16)in[i];
}

__global__ void k_transpose_f32_f16(const float* __restrict__ in,
                                    _Float16* __restrict__ out,
                                    int rows, int cols) {
  int i = blockIdx.x * blockDim.x + threadIdx.x;
  int stride = gridDim.x * blockDim.x;
  int total = rows * cols;
  for (; i < total; i += stride) {
    int r = i / cols, c = i % cols;
    out[(size_t)c * rows + r] = (_Float16)in[i];
  }
}

// ---------------------------------------------------------------------------
// WMMA GEMM: C(MxN) = A(MxK) * Bt(NxK)^T. Block = 256 threads = 8 waves,
// 128x128 output tile. The 128x32 Bt tile is DMA'd by the Tensor Data Mover
// into double-buffered LDS (wave 0 issues, TENSORcnt-pipelined) so the DMA of
// step s+1 overlaps the 8 WMMAs of step s.
// mode 1: f16 out, per-head (B,H,N,128) token-major (Q/K)
// mode 2: f16 out, per-head (B,H,128,N) transposed   (V)
// mode 3: f32 out + bias, row-major (final projection)
// ---------------------------------------------------------------------------
__global__ __launch_bounds__(256)
void k_gemm_wmma(const _Float16* __restrict__ A,
                 const _Float16* __restrict__ Bt,
                 float* __restrict__ outF, _Float16* __restrict__ outH,
                 const float* __restrict__ bias,
                 int M, int Nn, int K, int mode) {
  __shared__ __align__(16) _Float16 Bs[2][128 * 32];
  const int lane = threadIdx.x & 31;
  const int wave = threadIdx.x >> 5;
  const int m0 = blockIdx.x * 128 + wave * 16;
  const int n1 = blockIdx.y * 128;
  const int steps = K >> 5;

  const unsigned lds0 = (unsigned)(size_t)(void*)&Bs[0][0];
  const unsigned lds1 = (unsigned)(size_t)(void*)&Bs[1][0];
  const _Float16* btile = Bt + (size_t)n1 * K;  // 128 rows, stride K halves

  v8f acc[8] = {};

  if (wave == 0) tdm_load_tile_b16(lds0, btile, K, 128);

  for (int s = 0; s < steps; ++s) {
    const int buf = s & 1;
    if (wave == 0) {
      if (s + 1 < steps) {
        tdm_load_tile_b16(buf ? lds0 : lds1, btile + (size_t)(s + 1) * 32, K, 128);
        __builtin_amdgcn_s_wait_tensorcnt(1);  // buffer s done, s+1 in flight
      } else {
        __builtin_amdgcn_s_wait_tensorcnt(0);
      }
    }
    __syncthreads();  // tile visible to all waves

    const int k0 = s * 32;
    v16h a = load_op16(A + (size_t)m0 * K + k0, K);
    if (k0 + 32 < K)
      __builtin_prefetch(A + (size_t)m0 * K + k0 + 32, 0, 1);  // next A tile

#pragma unroll
    for (int t = 0; t < 8; ++t) {
      v16h b = load_op16(&Bs[buf][t * 16 * 32], 32);
      acc[t] = wmma32(a, b, acc[t]);
    }
    __syncthreads();  // all waves done with Bs[buf] before TDM reuses it
  }

  const int col = lane & 15;
  const int rb  = (lane >> 4) << 3;
#pragma unroll
  for (int t = 0; t < 8; ++t) {
    int ng = n1 + t * 16 + col;
#pragma unroll
    for (int i = 0; i < 8; ++i) {
      int mg = m0 + rb + i;
      float v = acc[t][i];
      if (mode == 1) {
        int bb = mg >> 10, tok = mg & 1023, hh = ng >> 7, dd = ng & 127;
        outH[((((size_t)bb * 12 + hh) << 10) + tok) * 128 + dd] = (_Float16)v;
      } else if (mode == 2) {
        int bb = mg >> 10, tok = mg & 1023, hh = ng >> 7, dd = ng & 127;
        outH[(((size_t)bb * 12 + hh) * 128 + dd) * 1024 + tok] = (_Float16)v;
      } else {
        outF[(size_t)mg * Nn + ng] = v + bias[ng];
      }
    }
  }
}

// ---------------------------------------------------------------------------
// Flash-style differential attention + LayerNorm.
// Block = 128 threads = 4 waves; each wave owns a 16-query tile of one (b,h).
// ---------------------------------------------------------------------------
__global__ __launch_bounds__(128)
void k_diff_attn(const _Float16* __restrict__ Q,   // (B,H,1024,128)
                 const _Float16* __restrict__ Kh,  // (B,H,1024,128)
                 const _Float16* __restrict__ Vt,  // (B,H,128,1024)
                 const float* __restrict__ lam_p,
                 const float* __restrict__ gamma,  // (128)
                 const float* __restrict__ beta,   // (128)
                 _Float16* __restrict__ attn) {    // (8192,1536)
  __shared__ __align__(16) _Float16 Ps[4][2][16 * 32];
  const int lane = threadIdx.x & 31;
  const int wave = threadIdx.x >> 5;
  const int bh = blockIdx.y;
  const int bb = bh / 12, hh = bh % 12;
  const int q0 = blockIdx.x * 64 + wave * 16;
  const float lam = lam_p[0];
  const float scale = 0.125f;  // D^-0.5, D=64

  const _Float16* qh = Q  + ((size_t)bh << 10) * 128;
  const _Float16* kh = Kh + ((size_t)bh << 10) * 128;
  const _Float16* vh = Vt + ((size_t)bh * 128) * 1024;

  v16h a1[2], a2[2];
  a1[0] = load_op16(qh + (size_t)q0 * 128 +  0, 128);
  a1[1] = load_op16(qh + (size_t)q0 * 128 + 32, 128);
  a2[0] = load_op16(qh + (size_t)q0 * 128 + 64, 128);
  a2[1] = load_op16(qh + (size_t)q0 * 128 + 96, 128);

  v8f O1[8] = {}, O2[8] = {};
  float m1[8], l1[8], m2[8], l2[8];
#pragma unroll
  for (int i = 0; i < 8; ++i) { m1[i] = m2[i] = -1e30f; l1[i] = l2[i] = 0.f; }

  const int col = lane & 15;
  const int rb  = (lane >> 4) << 3;
  _Float16* p1s = &Ps[wave][0][0];
  _Float16* p2s = &Ps[wave][1][0];

  for (int kk = 0; kk < 1024; kk += 32) {
    v8f s1a = {}, s1b = {}, s2a = {}, s2b = {};
    {
      v16h kb;
      kb = load_op16(kh + (size_t)kk * 128 +  0, 128);        s1a = wmma32(a1[0], kb, s1a);
      kb = load_op16(kh + (size_t)kk * 128 + 32, 128);        s1a = wmma32(a1[1], kb, s1a);
      kb = load_op16(kh + (size_t)(kk + 16) * 128 +  0, 128); s1b = wmma32(a1[0], kb, s1b);
      kb = load_op16(kh + (size_t)(kk + 16) * 128 + 32, 128); s1b = wmma32(a1[1], kb, s1b);
      kb = load_op16(kh + (size_t)kk * 128 + 64, 128);        s2a = wmma32(a2[0], kb, s2a);
      kb = load_op16(kh + (size_t)kk * 128 + 96, 128);        s2a = wmma32(a2[1], kb, s2a);
      kb = load_op16(kh + (size_t)(kk + 16) * 128 + 64, 128); s2b = wmma32(a2[0], kb, s2b);
      kb = load_op16(kh + (size_t)(kk + 16) * 128 + 96, 128); s2b = wmma32(a2[1], kb, s2b);
    }

    float f1[8], f2[8];
#pragma unroll
    for (int i = 0; i < 8; ++i) {
      // stream 1
      float xa = s1a[i] * scale, xb = s1b[i] * scale;
      float mn = fmaxf(m1[i], redmax16(fmaxf(xa, xb)));
      f1[i] = __expf(m1[i] - mn);
      float pa = __expf(xa - mn), pb = __expf(xb - mn);
      l1[i] = l1[i] * f1[i] + redsum16(pa + pb);
      m1[i] = mn;
      p1s[(rb + i) * 32 + col]      = (_Float16)pa;
      p1s[(rb + i) * 32 + col + 16] = (_Float16)pb;
      // stream 2
      xa = s2a[i] * scale; xb = s2b[i] * scale;
      mn = fmaxf(m2[i], redmax16(fmaxf(xa, xb)));
      f2[i] = __expf(m2[i] - mn);
      pa = __expf(xa - mn); pb = __expf(xb - mn);
      l2[i] = l2[i] * f2[i] + redsum16(pa + pb);
      m2[i] = mn;
      p2s[(rb + i) * 32 + col]      = (_Float16)pa;
      p2s[(rb + i) * 32 + col + 16] = (_Float16)pb;
    }

    v16h pa1 = load_op16(p1s, 32);   // same-wave LDS RAW: DScnt handled by compiler
    v16h pa2 = load_op16(p2s, 32);
#pragma unroll
    for (int t = 0; t < 8; ++t) {
      v16h vb = load_op16(vh + (size_t)(t * 16) * 1024 + kk, 1024);
#pragma unroll
      for (int i = 0; i < 8; ++i) {
        O1[t][i] = O1[t][i] * f1[i];
        O2[t][i] = O2[t][i] * f2[i];
      }
      O1[t] = wmma32(pa1, vb, O1[t]);
      O2[t] = wmma32(pa2, vb, O2[t]);
    }
  }

  // epilogue: normalize + differential combine
#pragma unroll
  for (int i = 0; i < 8; ++i) {
    float r1 = 1.f / l1[i], r2 = lam / l2[i];
#pragma unroll
    for (int t = 0; t < 8; ++t) O1[t][i] = O1[t][i] * r1 - O2[t][i] * r2;
  }
  // LayerNorm over 128 channels per row, *(1 - lambda_init)=0.8, store f16
#pragma unroll
  for (int i = 0; i < 8; ++i) {
    float s = 0.f, ss = 0.f;
#pragma unroll
    for (int t = 0; t < 8; ++t) { float v = O1[t][i]; s += v; ss += v * v; }
    s = redsum16(s); ss = redsum16(ss);
    float mu = s * (1.f / 128.f);
    float var = ss * (1.f / 128.f) - mu * mu;
    float rstd = rsqrtf(var + 1e-5f);
    int tok = q0 + rb + i;
    _Float16* dst = attn + ((size_t)bb * 1024 + tok) * 1536 + hh * 128;
#pragma unroll
    for (int t = 0; t < 8; ++t) {
      int dd = t * 16 + col;
      float v = (O1[t][i] - mu) * rstd * gamma[dd] + beta[dd];
      dst[dd] = (_Float16)(v * 0.8f);
    }
  }
}

// ---------------------------------------------------------------------------
extern "C" void kernel_launch(void* const* d_in, const int* in_sizes, int n_in,
                              void* d_out, int out_size, void* d_ws, size_t ws_size,
                              hipStream_t stream) {
  const float* x     = (const float*)d_in[0];
  const float* Wq    = (const float*)d_in[1];
  const float* Wk    = (const float*)d_in[2];
  const float* Wv    = (const float*)d_in[3];
  const float* lam   = (const float*)d_in[4];
  const float* gamma = (const float*)d_in[5];
  const float* beta  = (const float*)d_in[6];
  const float* Wp    = (const float*)d_in[7];
  const float* bp    = (const float*)d_in[8];
  float* out = (float*)d_out;

  const size_t XH = (size_t)8192 * 768;           // x f16
  const size_t WT = (size_t)1536 * 768;           // each transposed weight
  const size_t QK = (size_t)8 * 12 * 1024 * 128;  // q / k / vt per-head
  // workspace layout (halves)
  _Float16* xh  = (_Float16*)d_ws;
  _Float16* wqt = xh + XH;
  _Float16* wkt = wqt + WT;
  _Float16* wvt = wkt + WT;
  _Float16* wpt = wvt + WT;
  _Float16* q   = wpt + WT;
  _Float16* k   = q + QK;
  _Float16* vt  = k + QK;
  _Float16* at  = vt + QK;                        // (8192,1536) f16

  k_cvt_f32_f16<<<2048, 256, 0, stream>>>(x, xh, (int)XH);
  k_transpose_f32_f16<<<2048, 256, 0, stream>>>(Wq, wqt, 768, 1536);
  k_transpose_f32_f16<<<2048, 256, 0, stream>>>(Wk, wkt, 768, 1536);
  k_transpose_f32_f16<<<2048, 256, 0, stream>>>(Wv, wvt, 768, 1536);
  k_transpose_f32_f16<<<2048, 256, 0, stream>>>(Wp, wpt, 1536, 768);

  dim3 g1(64, 12);  // 8192/128 x 1536/128
  k_gemm_wmma<<<g1, 256, 0, stream>>>(xh, wqt, nullptr, q,  nullptr, 8192, 1536, 768, 1);
  k_gemm_wmma<<<g1, 256, 0, stream>>>(xh, wkt, nullptr, k,  nullptr, 8192, 1536, 768, 1);
  k_gemm_wmma<<<g1, 256, 0, stream>>>(xh, wvt, nullptr, vt, nullptr, 8192, 1536, 768, 2);

  dim3 g2(16, 96);  // 1024/64 query tiles x (B*H)
  k_diff_attn<<<g2, 128, 0, stream>>>(q, k, vt, lam, gamma, beta, at);

  dim3 g3(64, 6);   // 8192/128 x 768/128
  k_gemm_wmma<<<g3, 256, 0, stream>>>(at, wpt, out, nullptr, bp, 8192, 768, 1536, 3);
}
